// DotProductAttention_74062416052580
// MI455X (gfx1250) — matrix operand order repaired and checked by reading, not compiled, not run on hardware
//
#include <hip/hip_runtime.h>

typedef __attribute__((ext_vector_type(16))) _Float16 v16h;
typedef __attribute__((ext_vector_type(4)))  _Float16 v4h;
typedef __attribute__((ext_vector_type(8)))  float    v8f;
typedef __attribute__((ext_vector_type(4)))  int      v4i;
typedef __attribute__((ext_vector_type(8)))  int      v8i;
typedef __attribute__((ext_vector_type(4)))  unsigned int v4u;

#define ATT_B 8
#define ATT_L 2048
#define ATT_S 2048
#define ATT_D 128
#define BQ 64                    // query rows per block (4 waves x 16)
#define NW 4                     // waves per block
#define BSK 64                   // keys per S-iteration
#define SCALE_LOG2E (64.0f * 1.44269504088896340736f)  // softmax temp * log2(e)

#define AS1 __attribute__((address_space(1)))
#define AS3 __attribute__((address_space(3)))

#if defined(__gfx1250__) && __has_builtin(__builtin_amdgcn_global_load_async_to_lds_b128) && __has_builtin(__builtin_amdgcn_s_wait_asynccnt)
#define USE_ASYNC 1
#else
#define USE_ASYNC 0
#endif

#if defined(__gfx1250__) && __has_builtin(__builtin_amdgcn_tensor_load_to_lds) && __has_builtin(__builtin_amdgcn_s_wait_tensorcnt)
#define USE_TDM 1
#else
#define USE_TDM 0
#endif

__device__ __forceinline__ void copy_b128_to_lds(const _Float16* g, _Float16* l) {
#if USE_ASYNC
  __builtin_amdgcn_global_load_async_to_lds_b128((AS1 v4i*)g, (AS3 v4i*)l, 0, 0);
#else
  *(float4*)l = *(const float4*)g;
#endif
}

#if USE_TDM
// One TDM DMA of a 2D tile (tileW x tileH elements of 2 bytes, row stride in
// elements) from global to LDS.  D# layout per CDNA5 ISA 8.3/8.4.
__device__ __forceinline__ void tdm_load_2d(const _Float16* gsrc, _Float16* ldst,
                                            unsigned tileW, unsigned tileH,
                                            unsigned rowStride) {
  unsigned long long ga = (unsigned long long)(uintptr_t)gsrc;
  unsigned lds = (unsigned)(uintptr_t)(AS3 _Float16*)ldst;
  v4u g0;
  g0[0] = 1u;                                                // count=1, user D#
  g0[1] = lds;                                               // lds_addr
  g0[2] = (unsigned)(ga & 0xFFFFFFFFu);                      // global_addr[31:0]
  g0[3] = (unsigned)((ga >> 32) & 0x01FFFFFFu) | (2u << 30); // addr[56:32] | type=2
  v8i g1;
  g1[0] = (int)(1u << 16);                                   // data_size=1 (2B)
  g1[1] = (int)((tileW & 0xFFFFu) << 16);                    // tensor_dim0[15:0]
  g1[2] = (int)(((tileW >> 16) & 0xFFFFu) |                  // tensor_dim0[31:16]
                ((tileH & 0xFFFFu) << 16));                  // tensor_dim1[15:0]
  g1[3] = (int)(((tileH >> 16) & 0xFFFFu) |                  // tensor_dim1[31:16]
                ((tileW & 0xFFFFu) << 16));                  // tile_dim0
  g1[4] = (int)(tileH & 0xFFFFu);                            // tile_dim1 (tile_dim2=0)
  g1[5] = (int)rowStride;                                    // tensor_dim0_stride[31:0]
  g1[6] = 0;                                                 // stride0 hi | stride1 lo
  g1[7] = 0;
  v4i z4; z4[0] = z4[1] = z4[2] = z4[3] = 0;                 // 2D: groups 2/3 unused
  v8i z8;
  #pragma unroll
  for (int i = 0; i < 8; ++i) z8[i] = 0;
  __builtin_amdgcn_tensor_load_to_lds(g0, g1, z4, z4, z8, 0);
}
#endif

// ---------------- Kernel 1: L2-normalize Q,K -> f16; V -> f16 transposed ----
__global__ __launch_bounds__(256) void prep_norm_f16(
    const float* __restrict__ q, const float* __restrict__ k,
    const float* __restrict__ v,
    _Float16* __restrict__ qh, _Float16* __restrict__ kh,
    _Float16* __restrict__ vt)
{
  const int rows = ATT_B * ATT_L;
  int gw   = (blockIdx.x * blockDim.x + threadIdx.x) >> 5;  // global wave id
  int lane = threadIdx.x & 31;
  if (gw >= 3 * rows) return;

  if (gw < 2 * rows) {
    const float* src; _Float16* dst; int r = gw;
    if (r < rows) { src = q; dst = qh; }
    else          { r -= rows; src = k; dst = kh; }
    float4 x = ((const float4*)(src + (size_t)r * ATT_D))[lane];
    float ss = x.x*x.x + x.y*x.y + x.z*x.z + x.w*x.w;
    #pragma unroll
    for (int off = 16; off > 0; off >>= 1) ss += __shfl_xor(ss, off, 32);
    float inv = 1.0f / fmaxf(sqrtf(ss), 1e-12f);
    v4h h;
    h[0] = (_Float16)(x.x * inv);
    h[1] = (_Float16)(x.y * inv);
    h[2] = (_Float16)(x.z * inv);
    h[3] = (_Float16)(x.w * inv);
    *(v4h*)(dst + (size_t)r * ATT_D + lane * 4) = h;
  } else {
    int r = gw - 2 * rows;
    int b = r >> 11, srow = r & 2047;
    float4 x = ((const float4*)(v + (size_t)r * ATT_D))[lane];
    _Float16* base = vt + ((size_t)b * ATT_D + lane * 4) * ATT_S + srow;
    base[0 * ATT_S] = (_Float16)x.x;
    base[1 * ATT_S] = (_Float16)x.y;
    base[2 * ATT_S] = (_Float16)x.z;
    base[3 * ATT_S] = (_Float16)x.w;
  }
}

// ---------------- Kernel 2: flash attention, WMMA + TDM/async pipeline ------
__global__ __launch_bounds__(NW * 32) void flash_attn_f16(
    const _Float16* __restrict__ qh, const _Float16* __restrict__ kh,
    const _Float16* __restrict__ vt, float* __restrict__ out)
{
  __shared__ alignas(32) _Float16 ldsK [2][BSK * ATT_D];   // [buf][key][d]   2x16KB
  __shared__ alignas(32) _Float16 ldsVt[2][ATT_D * BSK];   // [buf][d][key]   2x16KB
  __shared__ alignas(32) _Float16 ldsP [NW * 16 * BSK];    // per-wave P      8KB

  const int tid   = threadIdx.x;
  const int lane  = tid & 31;
  const int wave  = tid >> 5;
  const int mlane = lane & 15;
  const int hi    = lane >> 4;          // half-wave group
  const int kb    = hi ? 8 : 0;         // A-layout K base for this half

  const int b     = blockIdx.x / (ATT_L / BQ);
  const int qtile = blockIdx.x % (ATT_L / BQ);
  const int qrow0 = qtile * BQ + wave * 16;

  const _Float16* Kb  = kh + (size_t)b * ATT_S * ATT_D;
  const _Float16* Vtb = vt + (size_t)b * ATT_D * ATT_S;

  // ---- Q A-fragments: 4 chunks of K=32 along D (contiguous k order: k = 16*hi + s)
  v16h afrag[4];
  const _Float16* qrow = qh + ((size_t)b * ATT_L + qrow0 + mlane) * ATT_D;
  #pragma unroll
  for (int c = 0; c < 4; ++c) {
    v16h t;
    ((float4*)&t)[0] = *(const float4*)(qrow + c * 32 + hi * 16);
    ((float4*)&t)[1] = *(const float4*)(qrow + c * 32 + hi * 16 + 8);
    afrag[c] = t;
  }

  v8f acc[8];
  #pragma unroll
  for (int d = 0; d < 8; ++d)
    #pragma unroll
    for (int i = 0; i < 8; ++i) acc[d][i] = 0.0f;

  float mrow[8], lrow[8];
  #pragma unroll
  for (int r2 = 0; r2 < 8; ++r2) { mrow[r2] = -__builtin_inff(); lrow[r2] = 0.0f; }

  _Float16* pw = ldsP + wave * 16 * BSK;

  auto issue_tile = [&](int s0, int buf) {
#if USE_TDM
    if (wave == 0) {
      // K tile: 64 rows x 128 elems, row stride 128 (contiguous)
      tdm_load_2d(Kb + (size_t)s0 * ATT_D, &ldsK[buf][0], ATT_D, BSK, ATT_D);
      // Vt tile: 128 rows x 64 elems, row stride 2048
      tdm_load_2d(Vtb + s0, &ldsVt[buf][0], BSK, ATT_D, ATT_S);
    }
#else
    const _Float16* gk = Kb + (size_t)s0 * ATT_D + tid * 64;
    _Float16* lk = &ldsK[buf][tid * 64];
    #pragma unroll
    for (int u = 0; u < 8; ++u) copy_b128_to_lds(gk + u * 8, lk + u * 8);
    const _Float16* gv = Vtb + (size_t)tid * ATT_S + s0;
    _Float16* lv = &ldsVt[buf][tid * BSK];
    #pragma unroll
    for (int u = 0; u < 8; ++u) copy_b128_to_lds(gv + u * 8, lv + u * 8);
#endif
  };

  auto wait_tiles = [&](bool more) {
#if USE_TDM
    if (wave == 0) {
      if (more) __builtin_amdgcn_s_wait_tensorcnt(2);   // cur tile's 2 DMAs done
      else      __builtin_amdgcn_s_wait_tensorcnt(0);
    }
#elif USE_ASYNC
    if (more) __builtin_amdgcn_s_wait_asynccnt(16);     // cur tile's 16 loads done
    else      __builtin_amdgcn_s_wait_asynccnt(0);
#endif
  };

  const int nIter = ATT_S / BSK;          // 32
  issue_tile(0, 0);

  for (int it = 0; it < nIter; ++it) {
    const int cur = it & 1;
    if (it > 0) __syncthreads();          // all waves done reading buf cur^1
    const bool more = (it + 1 < nIter);
    if (more) issue_tile((it + 1) * BSK, cur ^ 1);
    wait_tiles(more);                     // cur-tile transfers complete
    __syncthreads();                      // => cur-tile LDS visible to all

    const _Float16* Kt  = ldsK[cur];
    const _Float16* Vtt = ldsVt[cur];

    // ---- scores: 4 key-tiles x 4 D-chunks = 16 WMMAs
    v8f sc[4];
    #pragma unroll
    for (int t = 0; t < 4; ++t) {
      v8f c;
      #pragma unroll
      for (int i = 0; i < 8; ++i) c[i] = 0.0f;
      #pragma unroll
      for (int cc = 0; cc < 4; ++cc) {
        v16h bf = *(const v16h*)(Kt + (t * 16 + mlane) * ATT_D + cc * 32 + hi * 16);
        c = __builtin_amdgcn_wmma_f32_16x16x32_f16(false, afrag[cc], false, bf,
                                                   (short)0, c, false, false);
      }
      sc[t] = c;
    }

    // ---- online softmax (stats align with C-layout: reg r <-> row r+8*hi)
    float corr[8], sumr[8];
    #pragma unroll
    for (int r2 = 0; r2 < 8; ++r2) {
      float vmax = fmaxf(fmaxf(sc[0][r2], sc[1][r2]), fmaxf(sc[2][r2], sc[3][r2]));
      #pragma unroll
      for (int off = 8; off > 0; off >>= 1)
        vmax = fmaxf(vmax, __shfl_xor(vmax, off, 32));
      float mnew = fmaxf(mrow[r2], vmax);
      corr[r2] = exp2f((mrow[r2] - mnew) * SCALE_LOG2E);
      mrow[r2] = mnew;
      sumr[r2] = 0.0f;
    }
    #pragma unroll
    for (int t = 0; t < 4; ++t)
      #pragma unroll
      for (int r2 = 0; r2 < 8; ++r2) {
        float p = exp2f((sc[t][r2] - mrow[r2]) * SCALE_LOG2E);
        sumr[r2] += p;
        pw[(r2 + hi * 8) * BSK + t * 16 + mlane] = (_Float16)p;
      }
    #pragma unroll
    for (int r2 = 0; r2 < 8; ++r2) {
      float s = sumr[r2];
      #pragma unroll
      for (int off = 8; off > 0; off >>= 1) s += __shfl_xor(s, off, 32);
      lrow[r2] = lrow[r2] * corr[r2] + s;
    }
    #pragma unroll
    for (int d = 0; d < 8; ++d)
      #pragma unroll
      for (int r2 = 0; r2 < 8; ++r2) acc[d][r2] *= corr[r2];

    // ---- reload P in A-layout (wave-private LDS; DS in-order per wave)
    v16h pa[2];
    #pragma unroll
    for (int pc = 0; pc < 2; ++pc) {
      ((float4*)&pa[pc])[0] = *(const float4*)(pw + mlane * BSK + pc * 32 + hi * 16);
      ((float4*)&pa[pc])[1] = *(const float4*)(pw + mlane * BSK + pc * 32 + hi * 16 + 8);
    }

    // ---- PV: 8 D-tiles x 2 K-chunks = 16 WMMAs
    #pragma unroll
    for (int d = 0; d < 8; ++d)
      #pragma unroll
      for (int pc = 0; pc < 2; ++pc) {
        v16h bv = *(const v16h*)(Vtt + (d * 16 + mlane) * BSK + pc * 32 + hi * 16);
        acc[d] = __builtin_amdgcn_wmma_f32_16x16x32_f16(false, pa[pc], false, bv,
                                                        (short)0, acc[d], false, false);
      }
  }

  // ---- epilogue: divide by l, store fp32
  float* orow = out + ((size_t)b * ATT_L + qrow0) * ATT_D;
  #pragma unroll
  for (int r2 = 0; r2 < 8; ++r2) {
    float invl = 1.0f / lrow[r2];
    int m = r2 + hi * 8;
    #pragma unroll
    for (int d = 0; d < 8; ++d)
      orow[(size_t)m * ATT_D + d * 16 + mlane] = acc[d][r2] * invl;
  }
}

extern "C" void kernel_launch(void* const* d_in, const int* in_sizes, int n_in,
                              void* d_out, int out_size, void* d_ws, size_t ws_size,
                              hipStream_t stream) {
  const float* q = (const float*)d_in[0];
  const float* k = (const float*)d_in[1];
  const float* v = (const float*)d_in[2];
  float* out = (float*)d_out;

  const size_t elems = (size_t)ATT_B * ATT_L * ATT_D;   // 2,097,152
  _Float16* qh = (_Float16*)d_ws;
  _Float16* kh = qh + elems;
  _Float16* vt = kh + elems;                            // 12.6 MB total ws

  const int rows = ATT_B * ATT_L;
  const int totalThreads = 3 * rows * 32;               // one wave per row
  prep_norm_f16<<<(totalThreads + 255) / 256, 256, 0, stream>>>(q, k, v, qh, kh, vt);

  dim3 grid(ATT_B * (ATT_L / BQ));                      // 256 blocks
  flash_attn_f16<<<grid, NW * 32, 0, stream>>>(qh, kh, vt, out);
}
